// GATLayer_36232344109632
// MI455X (gfx1250) — compile-verified, matrix operand order checked by reading, not run on hardware
//
#include <hip/hip_runtime.h>
#include <hip/hip_bf16.h>

#define B_ 4
#define N_ 1024
#define IN_ 768
#define H_ 8
#define F_ 96
#define HF_ 768
#define ALPHA_ 0.2f
#define NEG_ -1e30f

typedef __attribute__((ext_vector_type(16))) __bf16 v16bf;
typedef __attribute__((ext_vector_type(16))) unsigned short v16u16;
typedef __attribute__((ext_vector_type(8))) float v8f;
typedef __attribute__((ext_vector_type(8))) unsigned v8u32;
typedef __attribute__((ext_vector_type(4))) unsigned v4u32;
typedef __attribute__((ext_vector_type(8))) int v8i32;
typedef __attribute__((ext_vector_type(4))) int v4i32;

#if defined(__HIP_DEVICE_COMPILE__) && __has_builtin(__builtin_amdgcn_tensor_load_to_lds)
#define USE_TDM 1
#else
#define USE_TDM 0
#endif

__device__ __forceinline__ unsigned short f32_to_bf16(float f) {
  unsigned int u = __builtin_bit_cast(unsigned int, f);
  unsigned int r = u + 0x7FFFu + ((u >> 16) & 1u);
  return (unsigned short)(r >> 16);
}

#if USE_TDM
// 2D tile DMA: global (bf16, row stride = stride0 elems) -> contiguous LDS.
// D# per CDNA5 ISA §8: group0 {count=1, lds_addr, global_addr, type=2},
// group1 {data_size=2B, tensor dims, tile dims, dim0 stride}. Groups 2/3 zero
// (2D tensor). clang-23 toolchain: 6-arg form with trailing int32x8 + cpol.
__device__ __forceinline__ void tdm_load_tile_2d(unsigned lds_addr, const void* gptr,
                                                 unsigned tensor_d0, unsigned tensor_d1,
                                                 unsigned stride0,
                                                 unsigned tile_d0, unsigned tile_d1) {
  unsigned long long ga = (unsigned long long)(uintptr_t)gptr;
  v4u32 g0;
  g0[0] = 1u;                                   // count = 1 valid descriptor
  g0[1] = lds_addr;                             // LDS byte address
  g0[2] = (unsigned)ga;                         // global_addr[31:0]
  g0[3] = (unsigned)(ga >> 32) | (2u << 30);    // global_addr[56:32] | type=2
  v8i32 g1;
  g1[0] = (int)(1u << 16);                      // data_size = 1 -> 2 bytes
  g1[1] = (int)(tensor_d0 << 16);               // tensor_dim0[15:0]
  g1[2] = (int)((tensor_d0 >> 16) | (tensor_d1 << 16));
  g1[3] = (int)((tensor_d1 >> 16) | (tile_d0 << 16));
  g1[4] = (int)(tile_d1 & 0xFFFFu);             // tile_dim1 (tile_dim2 = 0)
  g1[5] = (int)stride0;                         // tensor_dim0_stride[31:0]
  g1[6] = 0;
  g1[7] = 0;
  v4i32 z4 = {0, 0, 0, 0};
  v8i32 z8 = {0, 0, 0, 0, 0, 0, 0, 0};
  __builtin_amdgcn_tensor_load_to_lds(g0, g1, z4, z4, z8, 0);
}
#endif

// ---------------------------------------------------------------------------
// Kernel 0a: X (fp32) -> Xb (bf16), flat.
// ---------------------------------------------------------------------------
__global__ __launch_bounds__(256) void cvt_x_kernel(const float* __restrict__ X,
                                                    unsigned short* __restrict__ Xb) {
  int t = blockIdx.x * 256 + threadIdx.x;       // one float4 per thread
  const float4 v = ((const float4*)X)[t];
  unsigned lo = (unsigned)f32_to_bf16(v.x) | ((unsigned)f32_to_bf16(v.y) << 16);
  unsigned hi = (unsigned)f32_to_bf16(v.z) | ((unsigned)f32_to_bf16(v.w) << 16);
  ((uint2*)Xb)[t] = make_uint2(lo, hi);
}

// ---------------------------------------------------------------------------
// Kernel 0b: W (fp32, K x HF) -> Wt (bf16, HF x K transposed).
// ---------------------------------------------------------------------------
__global__ __launch_bounds__(256) void cvt_w_kernel(const float* __restrict__ W,
                                                    unsigned short* __restrict__ Wt) {
  int t = blockIdx.x * 256 + threadIdx.x;       // t = n*768 + k
  int n = t / IN_;
  int k = t - n * IN_;
  Wt[t] = f32_to_bf16(W[(size_t)k * HF_ + n]);
}

// ---------------------------------------------------------------------------
// Kernel 1: Wh = X @ W via bf16 WMMA; also f1 = Wh.a1, f2 = Wh.a2.
// grid: (32 m-blocks of 128 rows, 8 heads); block: 8 waves.
// W k-tile (96 f x 32 k, 64B rows) staged to LDS by the Tensor Data Mover.
// Wh stored transposed (b,h,f-major) so kernel 2 can also TDM its tiles.
// ---------------------------------------------------------------------------
__global__ __launch_bounds__(256) void gat_gemm_kernel(
    const unsigned short* __restrict__ Xb,    // (B*N, IN) bf16
    const unsigned short* __restrict__ Wt,    // (HF, IN) bf16 (transposed W)
    const float* __restrict__ avec,           // (2F)
    unsigned short* __restrict__ Whb_t,       // (B*H*F, N) bf16 transposed
    float* __restrict__ f1,                   // (B*H*N)
    float* __restrict__ f2)                   // (B*H*N)
{
  __shared__ unsigned short sm_w[F_ * 32];    // [f][k], rows 64B contiguous

  const int tid   = threadIdx.x;
  const int lane  = tid & 31;
  const int wave  = tid >> 5;
  const int h     = blockIdx.y;
  const int row0  = blockIdx.x * 128 + wave * 16;  // flat row (b*N + i)
  const int col   = lane & 15;
  const int halfk = (lane >> 4) * 8;

  v8f acc[6];
#pragma unroll
  for (int t = 0; t < 6; ++t) {
#pragma unroll
    for (int r = 0; r < 8; ++r) acc[t][r] = 0.f;
  }

  for (int k0 = 0; k0 < IN_; k0 += 32) {
#if USE_TDM
    if (wave == 0) {
      tdm_load_tile_2d((unsigned)(uintptr_t)sm_w,
                       Wt + (size_t)h * F_ * IN_ + k0,
                       IN_, F_, IN_, 32, F_);
    }
    __builtin_amdgcn_s_wait_tensorcnt(0);
#else
#pragma unroll
    for (int e = 0; e < 6; ++e) {             // 1536 dwords / 256 threads
      int lin = e * 256 + tid;                // lin = f*16 + kw
      int f   = lin >> 4;
      int kw  = lin & 15;
      ((unsigned*)sm_w)[lin] =
          *(const unsigned*)&Wt[((size_t)h * F_ + f) * IN_ + k0 + kw * 2];
    }
#endif
    __syncthreads();

    // A fragment: two contiguous 16B chunks per lane (K 0..7 and 16..23 rel).
    const unsigned short* xrow = Xb + (size_t)(row0 + col) * IN_ + k0 + halfk;
    uint4 alo = *(const uint4*)xrow;
    uint4 ahi = *(const uint4*)(xrow + 16);
    v8u32 a32;
    a32[0] = alo.x; a32[1] = alo.y; a32[2] = alo.z; a32[3] = alo.w;
    a32[4] = ahi.x; a32[5] = ahi.y; a32[6] = ahi.z; a32[7] = ahi.w;
    v16bf afrag = __builtin_bit_cast(v16bf, a32);

#pragma unroll
    for (int t = 0; t < 6; ++t) {
      const uint4* bp = (const uint4*)&sm_w[(t * 16 + col) * 32 + halfk];
      uint4 blo = bp[0];
      uint4 bhi = bp[2];
      v8u32 b32;
      b32[0] = blo.x; b32[1] = blo.y; b32[2] = blo.z; b32[3] = blo.w;
      b32[4] = bhi.x; b32[5] = bhi.y; b32[6] = bhi.z; b32[7] = bhi.w;
      v16bf bfrag = __builtin_bit_cast(v16bf, b32);
      acc[t] = __builtin_amdgcn_wmma_f32_16x16x32_bf16(
          false, afrag, false, bfrag, (short)0, acc[t], false, false);
    }
    __syncthreads();
  }

  // f1/f2 dot products in-register + transposed bf16 store (packed pairs).
  float a1w[6], a2w[6];
#pragma unroll
  for (int t = 0; t < 6; ++t) {
    a1w[t] = avec[t * 16 + col];
    a2w[t] = avec[F_ + t * 16 + col];
  }
  float p1[8], p2[8];
#pragma unroll
  for (int r = 0; r < 8; ++r) { p1[r] = 0.f; p2[r] = 0.f; }

  const int rowsel = (lane >> 4) * 8;
  const int b      = row0 >> 10;
  const int ibase  = (row0 & 1023) + rowsel;
#pragma unroll
  for (int t = 0; t < 6; ++t) {
    unsigned short* wp =
        Whb_t + (((size_t)((b * H_ + h) * F_ + t * 16 + col)) << 10) + ibase;
#pragma unroll
    for (int r = 0; r < 8; r += 2) {
      unsigned pk = (unsigned)f32_to_bf16(acc[t][r]) |
                    ((unsigned)f32_to_bf16(acc[t][r + 1]) << 16);
      *(unsigned*)(wp + r) = pk;
    }
#pragma unroll
    for (int r = 0; r < 8; ++r) {
      p1[r] += acc[t][r] * a1w[t];
      p2[r] += acc[t][r] * a2w[t];
    }
  }
#pragma unroll
  for (int off = 1; off < 16; off <<= 1) {
#pragma unroll
    for (int r = 0; r < 8; ++r) {
      p1[r] += __shfl_xor(p1[r], off, 32);
      p2[r] += __shfl_xor(p2[r], off, 32);
    }
  }
  if (col == 0) {
    size_t base = ((size_t)(b * H_ + h) << 10) + ibase;
#pragma unroll
    for (int r = 0; r < 8; ++r) {
      f1[base + r] = p1[r];
      f2[base + r] = p2[r];
    }
  }
}

// ---------------------------------------------------------------------------
// Kernel 2: fused leaky-relu + mask + softmax + (attn @ Wh) per (b,h).
// grid: (8 i-blocks, 8 heads, 4 batch); block: 8 waves x 16 rows x 96 cols.
// Scores are rank-1 (f1_i + f2_j): row max/denominator via cheap VALU prepass;
// normalized P built directly in WMMA A-layout. Wh tiles DMA'd by TDM.
// ---------------------------------------------------------------------------
__global__ __launch_bounds__(256) void gat_attn_kernel(
    const unsigned short* __restrict__ Whb_t,  // (B*H*F, N) bf16
    const float* __restrict__ f1,              // (B*H*N)
    const float* __restrict__ f2,              // (B*H*N)
    const int* __restrict__ mask,              // (B*N)
    float* __restrict__ out)                   // (B*N, HF)
{
  __shared__ float sm_f2[N_];
  __shared__ unsigned sm_maskbits[N_ / 32];
  __shared__ float sm_m[128];
  __shared__ float sm_l[128];
  __shared__ float sm_red[256];
  __shared__ unsigned short sm_tile[F_ * 32];  // [f][j], rows 64B contiguous

  const int tid  = threadIdx.x;
  const int lane = tid & 31;
  const int wave = tid >> 5;
  const int i0   = blockIdx.x * 128;
  const int h    = blockIdx.y;
  const int b    = blockIdx.z;
  const size_t bh = (size_t)(b * H_ + h) << 10;

#pragma unroll
  for (int e = 0; e < 4; ++e) sm_f2[e * 256 + tid] = f2[bh + e * 256 + tid];
  if (tid < 32) {
    unsigned bits = 0;
    const int* mp = mask + b * N_ + tid * 32;
    for (int j = 0; j < 32; ++j) bits |= (mp[j] != 0 ? (1u << j) : 0u);
    sm_maskbits[tid] = bits;
  }
  __syncthreads();

  // ---- prepass: per-row max m_i and denominator l_i ----
  {
    const int ir = tid & 127;
    const int halfj = tid >> 7;
    const float f1v = f1[bh + i0 + ir];
    float m = NEG_;
    for (int j = halfj * 512; j < halfj * 512 + 512; ++j) {
      if ((sm_maskbits[j >> 5] >> (j & 31)) & 1u) {
        float e = f1v + sm_f2[j];
        e = e >= 0.f ? e : ALPHA_ * e;
        m = fmaxf(m, e);
      }
    }
    sm_red[tid] = m;
    __syncthreads();
    if (tid < 128) sm_m[tid] = fmaxf(sm_red[tid], sm_red[tid + 128]);
    __syncthreads();
    const float mrow = sm_m[ir];
    float l = 0.f;
    for (int j = halfj * 512; j < halfj * 512 + 512; ++j) {
      if ((sm_maskbits[j >> 5] >> (j & 31)) & 1u) {
        float e = f1v + sm_f2[j];
        e = e >= 0.f ? e : ALPHA_ * e;
        l += __expf(e - mrow);
      }
    }
    sm_red[tid] = l;
    __syncthreads();
    if (tid < 128) sm_l[tid] = sm_red[tid] + sm_red[tid + 128];
    __syncthreads();
  }

  const int col   = lane & 15;
  const int halfk = (lane >> 4) * 8;
  const int myrow = wave * 16 + col;
  const float f1v  = f1[bh + i0 + myrow];
  const float mrow = sm_m[myrow];
  const float linv = 1.f / sm_l[myrow];

  v8f acc[6];
#pragma unroll
  for (int t = 0; t < 6; ++t) {
#pragma unroll
    for (int r = 0; r < 8; ++r) acc[t][r] = 0.f;
  }

  const int relk[8] = {0, 2, 4, 6, 16, 18, 20, 22};

  for (int js = 0; js < N_; js += 32) {
#if USE_TDM
    if (wave == 0) {
      tdm_load_tile_2d((unsigned)(uintptr_t)sm_tile,
                       Whb_t + ((size_t)(b * H_ + h) * F_ << 10) + js,
                       N_, F_, N_, 32, F_);
    }
    __builtin_amdgcn_s_wait_tensorcnt(0);
#else
#pragma unroll
    for (int e = 0; e < 6; ++e) {             // 1536 dwords / 256 threads
      int lin = e * 256 + tid;                // lin = f*16 + jw
      int f   = lin >> 4;
      int jw  = lin & 15;
      ((unsigned*)sm_tile)[lin] =
          *(const unsigned*)&Whb_t[(((size_t)(b * H_ + h) * F_ + f) << 10) +
                                   js + jw * 2];
    }
#endif
    __syncthreads();

    // P fragment in WMMA A-layout: p = exp(lrelu(f1+f2)-m)/l, masked -> 0.
    v16u16 pu;
#pragma unroll
    for (int tt = 0; tt < 8; ++tt) {
#pragma unroll
      for (int q = 0; q < 2; ++q) {
        int j = js + halfk + relk[tt] + q;
        float p = 0.f;
        if ((sm_maskbits[j >> 5] >> (j & 31)) & 1u) {
          float e = f1v + sm_f2[j];
          e = e >= 0.f ? e : ALPHA_ * e;
          p = __expf(e - mrow) * linv;
        }
        pu[2 * tt + q] = f32_to_bf16(p);
      }
    }
    v16bf pfrag = __builtin_bit_cast(v16bf, pu);

#pragma unroll
    for (int t = 0; t < 6; ++t) {
      const uint4* bp = (const uint4*)&sm_tile[(t * 16 + col) * 32 + halfk];
      uint4 blo = bp[0];
      uint4 bhi = bp[2];
      v8u32 b32;
      b32[0] = blo.x; b32[1] = blo.y; b32[2] = blo.z; b32[3] = blo.w;
      b32[4] = bhi.x; b32[5] = bhi.y; b32[6] = bhi.z; b32[7] = bhi.w;
      v16bf bfrag = __builtin_bit_cast(v16bf, b32);
      acc[t] = __builtin_amdgcn_wmma_f32_16x16x32_bf16(
          false, pfrag, false, bfrag, (short)0, acc[t], false, false);
    }
    __syncthreads();
  }

  // Store output tile (normalization already folded into P).
  const int rowsel = (lane >> 4) * 8;
#pragma unroll
  for (int t = 0; t < 6; ++t) {
#pragma unroll
    for (int r = 0; r < 8; ++r) {
      out[(size_t)(b * N_ + i0 + wave * 16 + rowsel + r) * HF_ +
          h * F_ + t * 16 + col] = acc[t][r];
    }
  }
}

// ---------------------------------------------------------------------------
extern "C" void kernel_launch(void* const* d_in, const int* in_sizes, int n_in,
                              void* d_out, int out_size, void* d_ws, size_t ws_size,
                              hipStream_t stream) {
  const float* X    = (const float*)d_in[0];   // node_features (B,N,IN)
  const int*   mask = (const int*)d_in[1];     // attention_mask (B,N)
  const float* W    = (const float*)d_in[2];   // (IN, H*F)
  const float* a    = (const float*)d_in[3];   // (2F)
  float* out = (float*)d_out;

  // Workspace: Xb bf16 | Wt bf16 | Whb_t bf16 | f1 | f2
  char* ws = (char*)d_ws;
  unsigned short* Xb    = (unsigned short*)(ws);                       // 6.29 MB
  unsigned short* Wt    = (unsigned short*)(ws + 6291456);             // 1.18 MB
  unsigned short* Whb_t = (unsigned short*)(ws + 6291456 + 1179648);   // 6.29 MB
  float* f1 = (float*)(ws + 6291456 + 1179648 + 6291456);
  float* f2 = (float*)(ws + 6291456 + 1179648 + 6291456 + 131072);

  cvt_x_kernel<<<dim3((B_ * N_ * IN_ / 4) / 256), 256, 0, stream>>>(X, Xb);
  cvt_w_kernel<<<dim3((IN_ * HF_) / 256), 256, 0, stream>>>(W, Wt);

  dim3 g1(32, 8, 1);
  gat_gemm_kernel<<<g1, 256, 0, stream>>>(Xb, Wt, a, Whb_t, f1, f2);

  dim3 g2(8, 8, 4);
  gat_attn_kernel<<<g2, 256, 0, stream>>>(Whb_t, f1, f2, mask, out);
}
